// TransformerEncoder_28690381537932
// MI455X (gfx1250) — compile-verified, matrix operand order checked by reading, not run on hardware
//
#include <hip/hip_runtime.h>
#include <hip/hip_bf16.h>

// Problem constants (match reference)
#define B_   4
#define S_   1024
#define D_   512
#define H_   8
#define L_   4
#define F_   2048
#define C_   16
#define DK_  64
#define EPS_ 1e-5f

typedef __attribute__((ext_vector_type(16))) __bf16       v16bf;
typedef __attribute__((ext_vector_type(8)))  float        v8f;
typedef __attribute__((ext_vector_type(4)))  unsigned int u32x4;
typedef unsigned short ushort_t;

union Frag { v16bf v; u32x4 q[2]; };

__device__ __forceinline__ unsigned int f2bf_bits(float f) {
  unsigned int u = __builtin_bit_cast(unsigned int, f);
  u += 0x7FFFu + ((u >> 16) & 1u);          // round-to-nearest-even
  return u >> 16;
}
__device__ __forceinline__ ushort_t f2bf(float f) { return (ushort_t)f2bf_bits(f); }

// ---------------------------------------------------------------------------
// Tiled transpose + f32->bf16 convert:  out[c*R + r] = bf16(in[r*C + c])
// (batched via blockIdx.z with element strides sIn/sOut); 256 threads.
// Used to pre-transpose all weights (and V) so every GEMM runs the NT path
// with plain pipelined b128 fragment loads.
// ---------------------------------------------------------------------------
__global__ __launch_bounds__(256) void transpose_bf16_kernel(
    const float* __restrict__ in, ushort_t* __restrict__ out,
    int R, int Cn, long sIn, long sOut)
{
  __shared__ float tile[32][33];
  in  += (long)blockIdx.z * sIn;
  out += (long)blockIdx.z * sOut;
  int c0 = blockIdx.x * 32, r0 = blockIdx.y * 32;
  int tx = threadIdx.x & 31, ty = threadIdx.x >> 5;   // 32 x 8
#pragma unroll
  for (int i = 0; i < 32; i += 8) {
    int r = r0 + ty + i, c = c0 + tx;
    tile[ty + i][tx] = (r < R && c < Cn) ? in[(long)r * Cn + c] : 0.f;
  }
  __syncthreads();
#pragma unroll
  for (int i = 0; i < 32; i += 8) {
    int r = r0 + tx;            // output minor index
    int c = c0 + ty + i;        // output row
    if (c < Cn && r < R) out[(long)c * R + r] = f2bf(tile[tx][ty + i]);
  }
}

// ---------------------------------------------------------------------------
// Batched WMMA GEMM (NT only):  C = scale*(A x B^T) + bias
//   A: bf16 row-major [M,K] (lda) ;  B: bf16 row-major [N,K] (ldb)
// No LDS, no barriers: fragments are loaded straight from global (the two
// b128 loads per fragment match the 16-bit WMMA operand layout exactly),
// letting the compiler software-pipeline loads across K-steps.
// 128 threads = 4 waves (2x2); wave tile = 32x32 -> 4 WMMAs per K-step.
// Partial tiles: branch-free row clamps (M and N), guarded stores.
// Optional dual bf16 store (Cbf) feeds the next GEMM without a convert pass.
// ---------------------------------------------------------------------------
#define BM 64
#define BN 64
#define BK 32

__global__ __launch_bounds__(128) void gemm_wmma_nt(
    const ushort_t* __restrict__ A, const ushort_t* __restrict__ Bm,
    const float* __restrict__ bias, float* __restrict__ C,
    ushort_t* __restrict__ Cbf,
    int M, int N, int K, int lda, int ldb, int ldc,
    long sAb, long sAh, long sBb, long sBh, long sCb, long sCh,
    int Hdim, float scale)
{
  const int z  = blockIdx.z;
  const int zb = z / Hdim, zh = z % Hdim;
  const long cOff = (long)zb * sCb + (long)zh * sCh;
  A  += (long)zb * sAb + (long)zh * sAh;
  Bm += (long)zb * sBb + (long)zh * sBh;
  C  += cOff;
  if (Cbf) Cbf += cOff;

  const int t    = threadIdx.x;
  const int lane = t & 31;
  const int w    = t >> 5;                  // wave 0..3
  const int m0   = blockIdx.y * BM + (w & 1) * 32;
  const int n0   = blockIdx.x * BN + (w >> 1) * 32;
  const int g    = lane >> 4;
  const int lr   = lane & 15;

  // branch-free clamps keep partial-tile loads in-bounds without exec masks
  int ar0 = m0 + lr;       if (ar0 > M - 1) ar0 = M - 1;
  int ar1 = m0 + 16 + lr;  if (ar1 > M - 1) ar1 = M - 1;
  int br0 = n0 + lr;       if (br0 > N - 1) br0 = N - 1;
  int br1 = n0 + 16 + lr;  if (br1 > N - 1) br1 = N - 1;
  const ushort_t* Ar0 = A  + (long)ar0 * lda + 8 * g;
  const ushort_t* Ar1 = A  + (long)ar1 * lda + 8 * g;
  const ushort_t* Br0 = Bm + (long)br0 * ldb + 8 * g;
  const ushort_t* Br1 = Bm + (long)br1 * ldb + 8 * g;

  v8f acc00 = {0.f,0.f,0.f,0.f,0.f,0.f,0.f,0.f};
  v8f acc01 = acc00, acc10 = acc00, acc11 = acc00;

  for (int k0 = 0; k0 < K; k0 += BK) {
    Frag fa0, fa1, fb0, fb1;
    fa0.q[0] = *(const u32x4*)(Ar0 + k0);
    fa0.q[1] = *(const u32x4*)(Ar0 + k0 + 16);
    fa1.q[0] = *(const u32x4*)(Ar1 + k0);
    fa1.q[1] = *(const u32x4*)(Ar1 + k0 + 16);
    fb0.q[0] = *(const u32x4*)(Br0 + k0);
    fb0.q[1] = *(const u32x4*)(Br0 + k0 + 16);
    fb1.q[0] = *(const u32x4*)(Br1 + k0);
    fb1.q[1] = *(const u32x4*)(Br1 + k0 + 16);
    acc00 = __builtin_amdgcn_wmma_f32_16x16x32_bf16(false, fa0.v, false, fb0.v,
                                                    (short)0, acc00, false, false);
    acc01 = __builtin_amdgcn_wmma_f32_16x16x32_bf16(false, fa0.v, false, fb1.v,
                                                    (short)0, acc01, false, false);
    acc10 = __builtin_amdgcn_wmma_f32_16x16x32_bf16(false, fa1.v, false, fb0.v,
                                                    (short)0, acc10, false, false);
    acc11 = __builtin_amdgcn_wmma_f32_16x16x32_bf16(false, fa1.v, false, fb1.v,
                                                    (short)0, acc11, false, false);
  }

  // store: lane -> N = lane%16 ; vgpr v -> M = v + 8*(lane/16)
  const int c0 = n0 + lr, c1 = n0 + 16 + lr;
  float b0 = 0.f, b1v = 0.f;
  if (bias) {
    if (c0 < N) b0  = bias[c0];
    if (c1 < N) b1v = bias[c1];
  }
#pragma unroll
  for (int vv = 0; vv < 8; ++vv) {
    int ma = m0 + vv + 8 * g;
    int mb = m0 + 16 + vv + 8 * g;
    if (ma < M) {
      if (c0 < N) {
        float o = acc00[vv] * scale + b0;
        C[(long)ma * ldc + c0] = o;
        if (Cbf) Cbf[(long)ma * ldc + c0] = f2bf(o);
      }
      if (c1 < N) {
        float o = acc01[vv] * scale + b1v;
        C[(long)ma * ldc + c1] = o;
        if (Cbf) Cbf[(long)ma * ldc + c1] = f2bf(o);
      }
    }
    if (mb < M) {
      if (c0 < N) {
        float o = acc10[vv] * scale + b0;
        C[(long)mb * ldc + c0] = o;
        if (Cbf) Cbf[(long)mb * ldc + c0] = f2bf(o);
      }
      if (c1 < N) {
        float o = acc11[vv] * scale + b1v;
        C[(long)mb * ldc + c1] = o;
        if (Cbf) Cbf[(long)mb * ldc + c1] = f2bf(o);
      }
    }
  }
}

// ---------------------------------------------------------------------------
// Elementwise / reduction kernels (f32 math, fused bf16 secondary outputs)
// ---------------------------------------------------------------------------
__global__ void embed_pe_kernel(const int* __restrict__ src,
                                const float* __restrict__ emb,
                                float* __restrict__ x, ushort_t* __restrict__ xbf,
                                int total)
{
  int i = blockIdx.x * blockDim.x + threadIdx.x;
  if (i >= total) return;
  int d  = i % D_;
  int bs = i / D_;
  int s  = bs % S_;
  int tok = src[bs];
  float v = emb[(long)tok * D_ + d] * 22.62741699796952f; // sqrt(512)
  int half = d >> 1;
  float freq = expf(-9.210340371976184f * (float)(2 * half) / (float)D_);
  float ang = (float)s * freq;
  v += (d & 1) ? cosf(ang) : sinf(ang);
  x[i] = v;
  xbf[i] = f2bf(v);
}

__global__ __launch_bounds__(256) void layernorm_kernel(
    const float* __restrict__ x, const float* __restrict__ g,
    const float* __restrict__ b, float* __restrict__ y,
    ushort_t* __restrict__ ybf, int ncols)
{
  long row = blockIdx.x;
  const float* xr = x + row * (long)ncols;
  float* yr = y + row * (long)ncols;
  ushort_t* yb = ybf + row * (long)ncols;
  int t = threadIdx.x;
  float s = 0.f, s2 = 0.f;
  for (int c = t; c < ncols; c += 256) { float v = xr[c]; s += v; s2 += v * v; }
  for (int off = 16; off > 0; off >>= 1) {
    s  += __shfl_down(s,  off, 32);
    s2 += __shfl_down(s2, off, 32);
  }
  __shared__ float ws1[8], ws2[8];
  int lane = t & 31, w = t >> 5;
  if (lane == 0) { ws1[w] = s; ws2[w] = s2; }
  __syncthreads();
  if (t == 0) {
    float a = 0.f, c2 = 0.f;
    for (int i = 0; i < 8; ++i) { a += ws1[i]; c2 += ws2[i]; }
    float mean = a / (float)ncols;
    float var  = c2 / (float)ncols - mean * mean;
    ws1[0] = mean;
    ws2[0] = rsqrtf(var + EPS_);
  }
  __syncthreads();
  float mean = ws1[0], rstd = ws2[0];
  for (int c = t; c < ncols; c += 256) {
    float o = (xr[c] - mean) * rstd * g[c] + b[c];
    yr[c] = o;
    yb[c] = f2bf(o);
  }
}

__global__ void add_bn_kernel(float* __restrict__ x, const float* __restrict__ y,
                              ushort_t* __restrict__ xbf,
                              const float* __restrict__ g, const float* __restrict__ b,
                              const float* __restrict__ mean, const float* __restrict__ var,
                              int ncols, int total)
{
  int i = blockIdx.x * blockDim.x + threadIdx.x;
  if (i >= total) return;
  int c = i % ncols;
  float v = x[i] + y[i];
  float o = (v - mean[c]) * rsqrtf(var[c] + EPS_) * g[c] + b[c];
  x[i] = o;
  xbf[i] = f2bf(o);
}

__global__ void gelu_bn_kernel(float* __restrict__ h, ushort_t* __restrict__ hbf,
                               const float* __restrict__ g, const float* __restrict__ b,
                               const float* __restrict__ mean, const float* __restrict__ var,
                               int ncols, int total)
{
  int i = blockIdx.x * blockDim.x + threadIdx.x;
  if (i >= total) return;
  float v = h[i];
  v = 0.5f * v * (1.f + erff(v * 0.70710678118654752f));   // exact gelu
  int c = i % ncols;
  float o = (v - mean[c]) * rsqrtf(var[c] + EPS_) * g[c] + b[c];
  h[i] = o;
  hbf[i] = f2bf(o);
}

__global__ void bn_gelu_kernel(float* __restrict__ z, ushort_t* __restrict__ zbf,
                               const float* __restrict__ g, const float* __restrict__ b,
                               const float* __restrict__ mean, const float* __restrict__ var,
                               int ncols, int total)
{
  int i = blockIdx.x * blockDim.x + threadIdx.x;
  if (i >= total) return;
  int c = i % ncols;
  float v = (z[i] - mean[c]) * rsqrtf(var[c] + EPS_) * g[c] + b[c];
  float o = 0.5f * v * (1.f + erff(v * 0.70710678118654752f));
  z[i] = o;
  zbf[i] = f2bf(o);
}

// masked softmax over last dim (S_), in place on f32 (d_out) + fused bf16 copy
__global__ __launch_bounds__(256) void softmax_mask_kernel(
    float* __restrict__ aw, ushort_t* __restrict__ awbf,
    const int* __restrict__ src)
{
  long r = blockIdx.x;                 // 0 .. B*H*S-1
  int b = (int)(r / ((long)H_ * S_));
  float* rowp = aw + r * (long)S_;
  ushort_t* rowb = awbf + r * (long)S_;
  const int* srcb = src + (long)b * S_;
  int t = threadIdx.x;

  float vals[4];
  float mx = -3.4e38f;
#pragma unroll
  for (int j = 0; j < 4; ++j) {
    int c = t + j * 256;
    float v = rowp[c];
    if (srcb[c] == 0) v = -1.0e9f;
    vals[j] = v;
    mx = fmaxf(mx, v);
  }
  for (int off = 16; off > 0; off >>= 1) mx = fmaxf(mx, __shfl_down(mx, off, 32));
  __shared__ float sm1[8], sm2[8];
  int lane = t & 31, w = t >> 5;
  if (lane == 0) sm1[w] = mx;
  __syncthreads();
  if (t == 0) {
    float m = sm1[0];
    for (int i = 1; i < 8; ++i) m = fmaxf(m, sm1[i]);
    sm1[0] = m;
  }
  __syncthreads();
  mx = sm1[0];

  float sum = 0.f;
#pragma unroll
  for (int j = 0; j < 4; ++j) { vals[j] = expf(vals[j] - mx); sum += vals[j]; }
  for (int off = 16; off > 0; off >>= 1) sum += __shfl_down(sum, off, 32);
  if (lane == 0) sm2[w] = sum;
  __syncthreads();
  if (t == 0) {
    float a = 0.f;
    for (int i = 0; i < 8; ++i) a += sm2[i];
    sm2[0] = 1.f / a;
  }
  __syncthreads();
  float inv = sm2[0];
#pragma unroll
  for (int j = 0; j < 4; ++j) {
    int c = t + j * 256;
    float o = vals[j] * inv;
    rowp[c] = o;
    rowb[c] = f2bf(o);
  }
}

// ---------------------------------------------------------------------------
// Host-side orchestration
// ---------------------------------------------------------------------------
static inline void launch_gemm(const ushort_t* A, const ushort_t* Bm,
                               const float* bias, float* C, ushort_t* Cbf,
                               int M, int N, int K, int lda, int ldb, int ldc,
                               long sAb, long sAh, long sBb, long sBh,
                               long sCb, long sCh, int Hdim, int batches,
                               float scale, hipStream_t st)
{
  dim3 grid((N + BN - 1) / BN, (M + BM - 1) / BM, batches);
  gemm_wmma_nt<<<grid, 128, 0, st>>>(A, Bm, bias, C, Cbf, M, N, K, lda, ldb, ldc,
                                     sAb, sAh, sBb, sBh, sCb, sCh, Hdim, scale);
}

static inline void launch_transpose(const float* in, ushort_t* out, int R, int Cn,
                                    long sIn, long sOut, int batches, hipStream_t st)
{
  dim3 grid((Cn + 31) / 32, (R + 31) / 32, batches);
  transpose_bf16_kernel<<<grid, 256, 0, st>>>(in, out, R, Cn, sIn, sOut);
}

extern "C" void kernel_launch(void* const* d_in, const int* in_sizes, int n_in,
                              void* d_out, int out_size, void* d_ws, size_t ws_size,
                              hipStream_t stream)
{
  (void)in_sizes; (void)n_in; (void)out_size; (void)ws_size;
  const int*   src       = (const int*)d_in[0];
  const float* emb       = (const float*)d_in[1];
  const float* attn_W    = (const float*)d_in[2];   // (L,4,D,D)
  const float* attn_b    = (const float*)d_in[3];   // (L,4,D)
  const float* ln_attn_g = (const float*)d_in[4];
  const float* ln_attn_b = (const float*)d_in[5];
  const float* W1        = (const float*)d_in[6];   // (L,D,F)
  const float* b1        = (const float*)d_in[7];
  const float* W2        = (const float*)d_in[8];   // (L,F,D)
  const float* b2        = (const float*)d_in[9];
  const float* ln_ffn_g  = (const float*)d_in[10];
  const float* ln_ffn_b  = (const float*)d_in[11];
  const float* bnff_g    = (const float*)d_in[12];
  const float* bnff_b    = (const float*)d_in[13];
  const float* bnff_mean = (const float*)d_in[14];
  const float* bnff_var  = (const float*)d_in[15];
  const float* bnd_g     = (const float*)d_in[16];  // (L,2,D)
  const float* bnd_b     = (const float*)d_in[17];
  const float* bnd_mean  = (const float*)d_in[18];
  const float* bnd_var   = (const float*)d_in[19];
  const float* fin_g     = (const float*)d_in[20];
  const float* fin_b     = (const float*)d_in[21];
  const float* pre_W     = (const float*)d_in[22];  // (D, D/2)
  const float* pre_b     = (const float*)d_in[23];
  const float* bnc_g     = (const float*)d_in[24];
  const float* bnc_b     = (const float*)d_in[25];
  const float* bnc_mean  = (const float*)d_in[26];
  const float* bnc_var   = (const float*)d_in[27];
  const float* fc_W      = (const float*)d_in[28];  // (D/2, C)
  const float* fc_b      = (const float*)d_in[29];

  const int  M    = B_ * S_;                  // 4096
  const long nBS  = (long)M * D_;             // 2,097,152
  const long nBSF = (long)M * F_;             // 8,388,608
  const long nAW  = (long)B_ * H_ * S_ * S_;  // 33,554,432 per layer

  // ---- workspace carve ----
  char* base = (char*)d_ws;
  float* x   = (float*)base;                 base += nBS  * 4;
  float* xn  = (float*)base;                 base += nBS  * 4;
  float* q   = (float*)base;                 base += nBS  * 4;
  float* k   = (float*)base;                 base += nBS  * 4;
  float* v   = (float*)base;                 base += nBS  * 4;
  float* ctx = (float*)base;                 base += nBS  * 4;
  float* h   = (float*)base;                 base += nBSF * 4;

  ushort_t* xbf   = (ushort_t*)base;         base += (nBS  + 64) * 2;
  ushort_t* xnbf  = (ushort_t*)base;         base += (nBS  + 64) * 2;
  ushort_t* qbf   = (ushort_t*)base;         base += (nBS  + 64) * 2;
  ushort_t* kbf   = (ushort_t*)base;         base += (nBS  + 64) * 2;
  ushort_t* vT    = (ushort_t*)base;         base += (nBS  + 64) * 2;   // [B,D,S]
  ushort_t* ctxbf = (ushort_t*)base;         base += (nBS  + 64) * 2;
  ushort_t* hbf   = (ushort_t*)base;         base += (nBSF + 64) * 2;
  ushort_t* awbf  = (ushort_t*)base;         base += (nAW  + 64) * 2;
  ushort_t* aWT   = (ushort_t*)base;         base += ((long)L_*4*D_*D_ + 64) * 2;
  ushort_t* w1T   = (ushort_t*)base;         base += ((long)L_*D_*F_   + 64) * 2;
  ushort_t* w2T   = (ushort_t*)base;         base += ((long)L_*F_*D_   + 64) * 2;
  ushort_t* preWT = (ushort_t*)base;         base += ((long)D_*(D_/2)  + 64) * 2;
  ushort_t* fcWT  = (ushort_t*)base;         base += ((long)(D_/2)*C_  + 64) * 2;
  ushort_t* zbf   = (ushort_t*)base;         base += (B_*(D_/2) + 64) * 2;

  float* out_f  = (float*)d_out;
  float* logits = out_f;                  // [B, C]
  float* attn0  = out_f + (long)B_ * C_;  // [L,B,H,S,S]

  const int totBSD = (int)nBS;
  const int totBSF = (int)nBSF;

  // ---- per-call weight transposes to bf16 [N,K] (NT operands) ----
  launch_transpose(attn_W, aWT, D_, D_, (long)D_*D_, (long)D_*D_, L_*4, stream);
  launch_transpose(W1,     w1T, D_, F_, (long)D_*F_, (long)F_*D_, L_,   stream);
  launch_transpose(W2,     w2T, F_, D_, (long)F_*D_, (long)D_*F_, L_,   stream);
  launch_transpose(pre_W,  preWT, D_, D_/2, 0, 0, 1, stream);
  launch_transpose(fc_W,   fcWT,  D_/2, C_, 0, 0, 1, stream);

  // x = emb[src]*sqrt(D) + PE  (+ fused bf16 copy for layer-0 K/V gemms)
  embed_pe_kernel<<<(totBSD + 255) / 256, 256, 0, stream>>>(src, emb, x, xbf, totBSD);

  const float scl = 0.125f;               // 1/sqrt(DK)
  for (int l = 0; l < L_; ++l) {
    const ushort_t* WqT = aWT + ((long)(l * 4 + 0)) * D_ * D_;
    const ushort_t* WkT = aWT + ((long)(l * 4 + 1)) * D_ * D_;
    const ushort_t* WvT = aWT + ((long)(l * 4 + 2)) * D_ * D_;
    const ushort_t* WoT = aWT + ((long)(l * 4 + 3)) * D_ * D_;
    const float* bq = attn_b + (long)(l * 4 + 0) * D_;
    const float* bk = attn_b + (long)(l * 4 + 1) * D_;
    const float* bv = attn_b + (long)(l * 4 + 2) * D_;
    const float* bo = attn_b + (long)(l * 4 + 3) * D_;

    // qn = LN(x) (fused bf16); q/k/v projections (q,k dual-store bf16)
    layernorm_kernel<<<M, 256, 0, stream>>>(x, ln_attn_g + (long)l * D_,
                                            ln_attn_b + (long)l * D_, xn, xnbf, D_);
    launch_gemm(xnbf, WqT, bq, q, qbf, M, D_, D_, D_, D_, D_, 0,0,0,0,0,0, 1, 1, 1.f, stream);
    launch_gemm(xbf,  WkT, bk, k, kbf, M, D_, D_, D_, D_, D_, 0,0,0,0,0,0, 1, 1, 1.f, stream);
    launch_gemm(xbf,  WvT, bv, v, nullptr, M, D_, D_, D_, D_, D_, 0,0,0,0,0,0, 1, 1, 1.f, stream);
    // vT[b][d][s] = v[b][s][d]  (per-batch transpose for ctx NT gemm)
    launch_transpose(v, vT, S_, D_, (long)S_*D_, (long)D_*S_, B_, stream);

    float* aw = attn0 + (long)l * nAW;      // [B,H,S,S] in d_out
    // scores = scale * Q K^T  (batched over B*H; K rows are already [n][k])
    launch_gemm(qbf, kbf, nullptr, aw, nullptr, S_, S_, DK_, D_, D_, S_,
                (long)S_ * D_, (long)DK_,
                (long)S_ * D_, (long)DK_,
                (long)H_ * S_ * S_, (long)S_ * S_,
                H_, B_ * H_, scl, stream);
    softmax_mask_kernel<<<B_ * H_ * S_, 256, 0, stream>>>(aw, awbf, src);
    // ctx = aw @ V : NT with B = vT rows (d-index), batched
    launch_gemm(awbf, vT, nullptr, ctx, ctxbf, S_, DK_, S_, S_, S_, D_,
                (long)H_ * S_ * S_, (long)S_ * S_,
                (long)D_ * S_, (long)DK_ * S_,
                (long)S_ * D_, (long)DK_,
                H_, B_ * H_, 1.f, stream);
    // proj = ctx@Wo+bo -> xn ; x = BN0(x + proj)  (fused bf16 x copy)
    launch_gemm(ctxbf, WoT, bo, xn, nullptr, M, D_, D_, D_, D_, D_, 0,0,0,0,0,0, 1, 1, 1.f, stream);
    add_bn_kernel<<<(totBSD + 255) / 256, 256, 0, stream>>>(
        x, xn, xbf,
        bnd_g + (long)(l * 2 + 0) * D_, bnd_b + (long)(l * 2 + 0) * D_,
        bnd_mean + (long)(l * 2 + 0) * D_, bnd_var + (long)(l * 2 + 0) * D_,
        D_, totBSD);

    // FFN
    layernorm_kernel<<<M, 256, 0, stream>>>(x, ln_ffn_g + (long)l * D_,
                                            ln_ffn_b + (long)l * D_, xn, xnbf, D_);
    launch_gemm(xnbf, w1T + (long)l * F_ * D_, b1 + (long)l * F_, h, nullptr,
                M, F_, D_, D_, D_, F_, 0,0,0,0,0,0, 1, 1, 1.f, stream);
    gelu_bn_kernel<<<(totBSF + 255) / 256, 256, 0, stream>>>(
        h, hbf, bnff_g + (long)l * F_, bnff_b + (long)l * F_,
        bnff_mean + (long)l * F_, bnff_var + (long)l * F_, F_, totBSF);
    launch_gemm(hbf, w2T + (long)l * D_ * F_, b2 + (long)l * D_, q, nullptr,
                M, D_, F_, F_, F_, D_, 0,0,0,0,0,0, 1, 1, 1.f, stream);
    add_bn_kernel<<<(totBSD + 255) / 256, 256, 0, stream>>>(
        x, q, xbf,
        bnd_g + (long)(l * 2 + 1) * D_, bnd_b + (long)(l * 2 + 1) * D_,
        bnd_mean + (long)(l * 2 + 1) * D_, bnd_var + (long)(l * 2 + 1) * D_,
        D_, totBSD);
  }

  // final LN, classifier head
  layernorm_kernel<<<M, 256, 0, stream>>>(x, fin_g, fin_b, xn, xnbf, D_);
  float* zbuf = ctx;                                    // [B, D/2]
  // cls rows = xn[b, 0, :] via lda = S*D ; z = cls @ pre_W + pre_b
  launch_gemm(xnbf, preWT, pre_b, zbuf, nullptr, B_, D_ / 2, D_,
              S_ * D_, D_, D_ / 2, 0,0,0,0,0,0, 1, 1, 1.f, stream);
  bn_gelu_kernel<<<(B_ * (D_ / 2) + 255) / 256, 256, 0, stream>>>(
      zbuf, zbf, bnc_g, bnc_b, bnc_mean, bnc_var, D_ / 2, B_ * (D_ / 2));
  // logits = z @ fc_W + fc_b  -> d_out[0:64]
  launch_gemm(zbf, fcWT, fc_b, logits, nullptr, B_, C_, D_ / 2,
              D_ / 2, D_ / 2, C_, 0,0,0,0,0,0, 1, 1, 1.f, stream);
}
